// GroupedQueryAttention_17214228922403
// MI455X (gfx1250) — matrix operand order repaired and checked by reading, not compiled, not run on hardware
//
#include <hip/hip_runtime.h>
#include <cstdint>
#include <cstddef>

// ---------------- constants ----------------
#define S_LEN    4096
#define EMBED    1024
#define N_HEADS  16
#define KVH      4
#define HDIM     64
#define QKV_N    1536   // 1024 Q + 256 K + 256 V

typedef __attribute__((ext_vector_type(16))) _Float16 v16h;
typedef __attribute__((ext_vector_type(8)))  float    v8f;

// ---------------- WMMA helpers ----------------
__device__ __forceinline__ v8f wmma_f16(v16h a, v16h b, v8f c) {
    return __builtin_amdgcn_wmma_f32_16x16x32_f16(
        false, a, false, b, (short)0, c, false, false);
}

union Frag16 { v16h v; uint4 q[2]; _Float16 h[16]; };

// A-matrix (16x32 f16) fragment from LDS.
// row_base points at LDS row (already includes lane%16 row select).
// ISA layout: lanes 0-15 -> K {0..7,16..23}, lanes 16-31 -> K {8..15,24..31}.
__device__ __forceinline__ v16h lds_a_frag(const _Float16* row_base, int koff, int lane) {
    Frag16 u;
    const _Float16* p = row_base + koff + ((lane >> 4) << 3);
    u.q[0] = *(const uint4*)(p);
    u.q[1] = *(const uint4*)(p + 16);
    return u.v;
}

// B-matrix (32x16 f16) fragment from LDS stored TRANSPOSED as Bt[n][k].
// ISA layout: lanes 0-15 -> K {0..7,16..23}, lanes 16-31 -> K {8..15,24..31}.
__device__ __forceinline__ v16h lds_b_frag(const _Float16* col_base, int koff, int lane) {
    Frag16 u;
    const _Float16* p = col_base + koff + ((lane >> 4) << 3);
    u.q[0] = *(const uint4*)(p);
    u.q[1] = *(const uint4*)(p + 16);
    return u.v;
}

// ---------------- CDNA5 async global->LDS copy (ASYNCcnt path) ----------
// Per-lane 16-byte copy, no VGPR data round-trip. LDS aperture keeps the
// byte offset in addr[31:0], so truncating the generic pointer is valid.
__device__ __forceinline__ void async_ld16(void* lds, const void* g) {
    asm volatile("global_load_async_to_lds_b128 %0, %1, off"
                 :
                 : "v"((uint32_t)(uintptr_t)lds), "v"((uint64_t)(uintptr_t)g)
                 : "memory");
}
__device__ __forceinline__ void async_wait0() {
    asm volatile("s_wait_asynccnt 0x0" ::: "memory");
}

// ======================================================================
// Kernel 1: QKV projection.  qkvf[s][0:1536] = seq @ [W_q | W_kv]
// 128x128 tile per block, 256 threads (8 waves), wave -> 32x64 region.
// ======================================================================
__global__ __launch_bounds__(256) void k_gemm_qkv(
    const float* __restrict__ seq, const float* __restrict__ Wq,
    const float* __restrict__ Wkv, float* __restrict__ qkvf)
{
    __shared__ __align__(16) _Float16 As[128][40];
    __shared__ __align__(16) _Float16 Bt[128][40];

    const int m0 = blockIdx.y * 128;
    const int n0 = blockIdx.x * 128;
    const float* B; int ldb, bn0;
    if (n0 < 1024) { B = Wq;  ldb = 1024; bn0 = n0; }
    else           { B = Wkv; ldb = 512;  bn0 = n0 - 1024; }

    const int tid = threadIdx.x, lane = tid & 31, w = tid >> 5;
    const int wm = w & 3, wn = w >> 2;          // 4x2 wave grid

    v8f acc[8];
#pragma unroll
    for (int i = 0; i < 8; ++i)
        acc[i] = (v8f){0.f,0.f,0.f,0.f,0.f,0.f,0.f,0.f};

    for (int kc = 0; kc < EMBED; kc += 32) {
        // stage A tile (128x32 fp32 -> f16, vectorized LDS store)
        {
            const int row = tid >> 1, half = tid & 1;
            const float* src = seq + (size_t)(m0 + row) * EMBED + kc + half * 16;
            float buf[16];
            *(float4*)(&buf[0])  = *(const float4*)(src);
            *(float4*)(&buf[4])  = *(const float4*)(src + 4);
            *(float4*)(&buf[8])  = *(const float4*)(src + 8);
            *(float4*)(&buf[12]) = *(const float4*)(src + 12);
            Frag16 u;
#pragma unroll
            for (int j = 0; j < 16; ++j) u.h[j] = (_Float16)buf[j];
            *(uint4*)&As[row][half * 16]     = u.q[0];
            *(uint4*)&As[row][half * 16 + 8] = u.q[1];
        }
        // stage B tile: column-wise reads (coalesced across lanes),
        // contiguous transposed LDS store Bt[n][k].
        {
            const int n = tid & 127, kh = tid >> 7;   // n 0..127, k half 0/1
            float buf[16];
#pragma unroll
            for (int j = 0; j < 16; ++j)
                buf[j] = B[(size_t)(kc + kh * 16 + j) * ldb + bn0 + n];
            Frag16 u;
#pragma unroll
            for (int j = 0; j < 16; ++j) u.h[j] = (_Float16)buf[j];
            *(uint4*)&Bt[n][kh * 16]     = u.q[0];
            *(uint4*)&Bt[n][kh * 16 + 8] = u.q[1];
        }
        __syncthreads();

        v16h a0 = lds_a_frag(&As[wm * 32 +      (lane & 15)][0], 0, lane);
        v16h a1 = lds_a_frag(&As[wm * 32 + 16 + (lane & 15)][0], 0, lane);
#pragma unroll
        for (int sn = 0; sn < 4; ++sn) {
            v16h b = lds_b_frag(&Bt[wn * 64 + sn * 16 + (lane & 15)][0], 0, lane);
            acc[sn]     = wmma_f16(a0, b, acc[sn]);
            acc[4 + sn] = wmma_f16(a1, b, acc[4 + sn]);
        }
        __syncthreads();
    }

#pragma unroll
    for (int sm = 0; sm < 2; ++sm)
#pragma unroll
        for (int sn = 0; sn < 4; ++sn) {
            v8f a = acc[sm * 4 + sn];
            const int col = n0 + wn * 64 + sn * 16 + (lane & 15);
#pragma unroll
            for (int r = 0; r < 8; ++r) {
                const int row = m0 + wm * 32 + sm * 16 + r + ((lane >> 4) << 3);
                qkvf[(size_t)row * QKV_N + col] = a[r];
            }
        }
}

// ======================================================================
// Kernel 2: RoPE + pack to f16 (head-major) + write fp32 K/V cache.
// V is written TRANSPOSED ([kvh][d][s]) so attention can async-copy it.
// ======================================================================
__global__ __launch_bounds__(256) void k_rope(
    const float* __restrict__ qkvf,
    _Float16* __restrict__ Q16, _Float16* __restrict__ K16,
    _Float16* __restrict__ V16t,
    float* __restrict__ kcache, float* __restrict__ vcache)
{
    const int idx = blockIdx.x * 256 + threadIdx.x;
    const int QP = S_LEN * N_HEADS * 32;   // Q rope pairs
    const int KP = S_LEN * KVH * 32;       // K rope pairs (== V pairs)

    if (idx < QP) {
        const int j = idx & 31, t = idx >> 5;
        const int h = t & (N_HEADS - 1), s = t >> 4;
        const size_t src = (size_t)s * QKV_N + h * HDIM + j;
        const float x1 = qkvf[src], x2 = qkvf[src + 32];
        const float inv = __powf(10000.0f, -(float)(2 * j) * (1.0f / HDIM));
        float sn, cs; __sincosf((float)s * inv, &sn, &cs);
        const size_t base = ((size_t)h * S_LEN + s) * HDIM + j;
        Q16[base]      = (_Float16)(x1 * cs - x2 * sn);
        Q16[base + 32] = (_Float16)(x2 * cs + x1 * sn);
    } else if (idx < QP + KP) {
        const int i2 = idx - QP;
        const int j = i2 & 31, t = i2 >> 5;
        const int kvh = t & 3, s = t >> 2;
        const size_t src = (size_t)s * QKV_N + 1024 + kvh * HDIM + j;
        const float x1 = qkvf[src], x2 = qkvf[src + 32];
        const float inv = __powf(10000.0f, -(float)(2 * j) * (1.0f / HDIM));
        float sn, cs; __sincosf((float)s * inv, &sn, &cs);
        const float o1 = x1 * cs - x2 * sn, o2 = x2 * cs + x1 * sn;
        const size_t base = ((size_t)kvh * S_LEN + s) * HDIM + j;
        K16[base] = (_Float16)o1;  K16[base + 32] = (_Float16)o2;
        kcache[base] = o1;         kcache[base + 32] = o2;
    } else if (idx < QP + 2 * KP) {
        const int i2 = idx - QP - KP;
        const int j = i2 & 31, t = i2 >> 5;
        const int kvh = t & 3, s = t >> 2;
        const size_t src = (size_t)s * QKV_N + 1280 + kvh * HDIM + j;
        const float x1 = qkvf[src], x2 = qkvf[src + 32];
        // transposed f16 V: [kvh][d][s]
        V16t[((size_t)kvh * HDIM + j)      * S_LEN + s] = (_Float16)x1;
        V16t[((size_t)kvh * HDIM + j + 32) * S_LEN + s] = (_Float16)x2;
        // fp32 cache stays [kvh][s][d]
        const size_t base = ((size_t)kvh * S_LEN + s) * HDIM + j;
        vcache[base] = x1;  vcache[base + 32] = x2;
    }
}

// ======================================================================
// Kernel 3: causal flash attention, WMMA for QK^T and P*V.
// block = (query tile of 128 rows, head). 8 waves x 16 query rows.
// Q/K/V tiles staged with global_load_async_to_lds_b128.
// ======================================================================
__global__ __launch_bounds__(256) void k_attn(
    const _Float16* __restrict__ Q16, const _Float16* __restrict__ K16,
    const _Float16* __restrict__ V16t, _Float16* __restrict__ Ab)
{
    __shared__ __align__(16) _Float16 Qs[128][72];
    __shared__ __align__(16) _Float16 Ks[64][72];
    __shared__ __align__(16) _Float16 Vt[64][72];     // [d][kv]
    __shared__ __align__(16) _Float16 Ps[8][16][72];  // per-wave P tile

    const int qt = blockIdx.x, h = blockIdx.y;
    const int q0 = qt * 128, kvh = h >> 2;
    const int tid = threadIdx.x, lane = tid & 31, w = tid >> 5;

    // stage Q tile (128 x 64 f16) via async copies
    {
        const int row = tid >> 1, half = tid & 1;
        const _Float16* g = Q16 + ((size_t)h * S_LEN + q0 + row) * HDIM + half * 32;
#pragma unroll
        for (int j = 0; j < 4; ++j)
            async_ld16(&Qs[row][half * 32 + j * 8], g + j * 8);
    }
    async_wait0();
    __syncthreads();

    const v16h aq0 = lds_a_frag(&Qs[w * 16 + (lane & 15)][0], 0,  lane);
    const v16h aq1 = lds_a_frag(&Qs[w * 16 + (lane & 15)][0], 32, lane);

    v8f O[4];
#pragma unroll
    for (int i = 0; i < 4; ++i)
        O[i] = (v8f){0.f,0.f,0.f,0.f,0.f,0.f,0.f,0.f};
    float m_i[8], l_i[8];
#pragma unroll
    for (int r = 0; r < 8; ++r) { m_i[r] = -1e30f; l_i[r] = 0.f; }

    const int ntiles = (q0 >> 6) + 2;   // causal: only KV tiles with k0 <= q0+127
    for (int t = 0; t < ntiles; ++t) {
        const int k0 = t << 6;
        __syncthreads();                 // all waves done reading previous tile
        // stage K tile [kv][d] and V tile [d][kv] (pre-transposed in memory)
        {
            const int row = tid >> 2, qd = tid & 3;   // row: kv for K, d for V
            const _Float16* gk = K16  + ((size_t)kvh * S_LEN + k0 + row) * HDIM + qd * 16;
            async_ld16(&Ks[row][qd * 16],     gk);
            async_ld16(&Ks[row][qd * 16 + 8], gk + 8);
            const _Float16* gv = V16t + ((size_t)kvh * HDIM + row) * S_LEN + k0 + qd * 16;
            async_ld16(&Vt[row][qd * 16],     gv);
            async_ld16(&Vt[row][qd * 16 + 8], gv + 8);
        }
        async_wait0();
        __syncthreads();

        // S = Q K^T  (4 subtiles of 16 cols, contraction 64 = 2 wmma)
        v8f sf[4];
#pragma unroll
        for (int nt = 0; nt < 4; ++nt) {
            v8f c = (v8f){0.f,0.f,0.f,0.f,0.f,0.f,0.f,0.f};
            c = wmma_f16(aq0, lds_b_frag(&Ks[nt * 16 + (lane & 15)][0], 0,  lane), c);
            c = wmma_f16(aq1, lds_b_frag(&Ks[nt * 16 + (lane & 15)][0], 32, lane), c);
            sf[nt] = c;
        }

        // scale + causal mask
        const int rowb = q0 + w * 16 + ((lane >> 4) << 3);
#pragma unroll
        for (int nt = 0; nt < 4; ++nt) {
            const int col = k0 + nt * 16 + (lane & 15);
#pragma unroll
            for (int r = 0; r < 8; ++r) {
                const float v = sf[nt][r] * 0.125f;   // 1/sqrt(64)
                sf[nt][r] = (col > rowb + r) ? -1e30f : v;
            }
        }

        // row max (reduce across 16 lanes of the half-wave)
        float mrow[8];
#pragma unroll
        for (int r = 0; r < 8; ++r)
            mrow[r] = fmaxf(fmaxf(sf[0][r], sf[1][r]), fmaxf(sf[2][r], sf[3][r]));
#pragma unroll
        for (int off = 1; off < 16; off <<= 1)
#pragma unroll
            for (int r = 0; r < 8; ++r)
                mrow[r] = fmaxf(mrow[r], __shfl_xor(mrow[r], off, 32));

        // online softmax update
        float psum[8];
#pragma unroll
        for (int r = 0; r < 8; ++r) {
            const float mnew  = fmaxf(m_i[r], mrow[r]);
            const float alpha = __expf(m_i[r] - mnew);
            m_i[r] = mnew;
            float srow = 0.f;
#pragma unroll
            for (int nt = 0; nt < 4; ++nt) {
                const float p = __expf(sf[nt][r] - mnew);
                sf[nt][r] = p;
                srow += p;
            }
            psum[r] = srow;
            l_i[r] *= alpha;
#pragma unroll
            for (int dn = 0; dn < 4; ++dn) O[dn][r] *= alpha;
        }
#pragma unroll
        for (int off = 1; off < 16; off <<= 1)
#pragma unroll
            for (int r = 0; r < 8; ++r)
                psum[r] += __shfl_xor(psum[r], off, 32);
#pragma unroll
        for (int r = 0; r < 8; ++r) l_i[r] += psum[r];

        // P: C-layout -> A-layout through per-wave LDS
#pragma unroll
        for (int nt = 0; nt < 4; ++nt)
#pragma unroll
            for (int r = 0; r < 8; ++r)
                Ps[w][r + ((lane >> 4) << 3)][nt * 16 + (lane & 15)] = (_Float16)sf[nt][r];

        const v16h p0 = lds_a_frag(&Ps[w][(lane & 15)][0], 0,  lane);
        const v16h p1 = lds_a_frag(&Ps[w][(lane & 15)][0], 32, lane);
#pragma unroll
        for (int dn = 0; dn < 4; ++dn) {
            O[dn] = wmma_f16(p0, lds_b_frag(&Vt[dn * 16 + (lane & 15)][0], 0,  lane), O[dn]);
            O[dn] = wmma_f16(p1, lds_b_frag(&Vt[dn * 16 + (lane & 15)][0], 32, lane), O[dn]);
        }
    }

    // normalize and write attention output as f16 [s][h*64 + d]
    float inv_l[8];
#pragma unroll
    for (int r = 0; r < 8; ++r) inv_l[r] = 1.0f / l_i[r];
#pragma unroll
    for (int dn = 0; dn < 4; ++dn) {
        const int col = h * HDIM + dn * 16 + (lane & 15);
#pragma unroll
        for (int r = 0; r < 8; ++r) {
            const int s = q0 + w * 16 + r + ((lane >> 4) << 3);
            Ab[(size_t)s * EMBED + col] = (_Float16)(O[dn][r] * inv_l[r]);
        }
    }
}

// ======================================================================
// Kernel 4: output projection.  out = Ab(f16) @ W_out + b_out  (fp32 out)
// ======================================================================
__global__ __launch_bounds__(256) void k_gemm_out(
    const _Float16* __restrict__ Ab, const float* __restrict__ Wout,
    const float* __restrict__ bias, float* __restrict__ out)
{
    __shared__ __align__(16) _Float16 As[128][40];
    __shared__ __align__(16) _Float16 Bt[128][40];

    const int m0 = blockIdx.y * 128;
    const int n0 = blockIdx.x * 128;
    const int tid = threadIdx.x, lane = tid & 31, w = tid >> 5;
    const int wm = w & 3, wn = w >> 2;

    v8f acc[8];
#pragma unroll
    for (int i = 0; i < 8; ++i)
        acc[i] = (v8f){0.f,0.f,0.f,0.f,0.f,0.f,0.f,0.f};

    for (int kc = 0; kc < EMBED; kc += 32) {
        {   // stage A (already f16): async copy 128x32 halves
            const int row = tid >> 1, half = tid & 1;
            const _Float16* g = Ab + (size_t)(m0 + row) * EMBED + kc + half * 16;
            async_ld16(&As[row][half * 16],     g);
            async_ld16(&As[row][half * 16 + 8], g + 8);
        }
        {   // stage B transposed: column-wise coalesced reads, vector LDS store
            const int n = tid & 127, kh = tid >> 7;
            float buf[16];
#pragma unroll
            for (int j = 0; j < 16; ++j)
                buf[j] = Wout[(size_t)(kc + kh * 16 + j) * EMBED + n0 + n];
            Frag16 u;
#pragma unroll
            for (int j = 0; j < 16; ++j) u.h[j] = (_Float16)buf[j];
            *(uint4*)&Bt[n][kh * 16]     = u.q[0];
            *(uint4*)&Bt[n][kh * 16 + 8] = u.q[1];
        }
        async_wait0();
        __syncthreads();

        v16h a0 = lds_a_frag(&As[wm * 32 +      (lane & 15)][0], 0, lane);
        v16h a1 = lds_a_frag(&As[wm * 32 + 16 + (lane & 15)][0], 0, lane);
#pragma unroll
        for (int sn = 0; sn < 4; ++sn) {
            v16h b = lds_b_frag(&Bt[wn * 64 + sn * 16 + (lane & 15)][0], 0, lane);
            acc[sn]     = wmma_f16(a0, b, acc[sn]);
            acc[4 + sn] = wmma_f16(a1, b, acc[4 + sn]);
        }
        __syncthreads();
    }

#pragma unroll
    for (int sm = 0; sm < 2; ++sm)
#pragma unroll
        for (int sn = 0; sn < 4; ++sn) {
            v8f a = acc[sm * 4 + sn];
            const int col = n0 + wn * 64 + sn * 16 + (lane & 15);
            const float bb = bias[col];
#pragma unroll
            for (int r = 0; r < 8; ++r) {
                const int row = m0 + wm * 32 + sm * 16 + r + ((lane >> 4) << 3);
                out[(size_t)row * EMBED + col] = a[r] + bb;
            }
        }
}

// ======================================================================
// launcher
// ======================================================================
extern "C" void kernel_launch(void* const* d_in, const int* in_sizes, int n_in,
                              void* d_out, int out_size, void* d_ws, size_t ws_size,
                              hipStream_t stream) {
    (void)in_sizes; (void)n_in; (void)out_size; (void)ws_size;

    const float* seq  = (const float*)d_in[0];
    const float* W_q  = (const float*)d_in[1];
    const float* W_kv = (const float*)d_in[2];
    const float* W_out= (const float*)d_in[3];
    const float* b_out= (const float*)d_in[4];

    float* out    = (float*)d_out;                         // [S, E]
    float* kcache = out + (size_t)S_LEN * EMBED;           // [KVH, S, D]
    float* vcache = kcache + (size_t)KVH * S_LEN * HDIM;   // [KVH, S, D]

    char* ws = (char*)d_ws;
    float*    qkvf = (float*)ws;                              // S*1536 fp32 = 25165824 B
    _Float16* Q16  = (_Float16*)(ws + 25165824);              // H*S*64  f16 =  8388608 B
    _Float16* K16  = (_Float16*)(ws + 33554432);              // KVH*S*64 f16 = 2097152 B
    _Float16* V16t = (_Float16*)(ws + 35651584);              // KVH*64*S f16 = 2097152 B
    _Float16* Ab   = (_Float16*)(ws + 37748736);              // S*E f16     = 8388608 B

    // 1) QKV projection
    k_gemm_qkv<<<dim3(QKV_N / 128, S_LEN / 128), 256, 0, stream>>>(seq, W_q, W_kv, qkvf);
    // 2) RoPE + pack + KV cache (V emitted transposed for async staging)
    {
        const int total = S_LEN * 32 * (N_HEADS + 2 * KVH);
        k_rope<<<(total + 255) / 256, 256, 0, stream>>>(qkvf, Q16, K16, V16t, kcache, vcache);
    }
    // 3) causal flash attention
    k_attn<<<dim3(S_LEN / 128, N_HEADS), 256, 0, stream>>>(Q16, K16, V16t, Ab);
    // 4) output projection + bias
    k_gemm_out<<<dim3(EMBED / 128, S_LEN / 128), 256, 0, stream>>>(Ab, W_out, b_out, out);
}
